// FactoredHmmLm_77249281786385
// MI455X (gfx1250) — matrix operand order repaired and checked by reading, not compiled, not run on hardware
//
#include <hip/hip_runtime.h>

// Problem constants (from reference)
#define Hdim 256
#define NCc  128
#define SPWc 64
#define CS   8192   // NC*SPW
#define VV   10000
#define BB   16
#define TT   256
#define NEGF (-1e30f)

typedef __attribute__((ext_vector_type(16))) __bf16 v16bf;
typedef __attribute__((ext_vector_type(8)))  float  v8f;

// ---------------------------------------------------------------------------
// WMMA fragment loaders for D = A * B^T, A (M x K) row-major, Bt (N x K)
// row-major, bf16, K multiple of 32.
// A 16x32 bf16 layout: lane m=lane&15, khalf=(lane>>4)*16,
//   elems e -> K=khalf+e (contiguous, matching B).
// B 32x16 bf16 layout used here: lane n=lane&15, khalf=(lane>>4)*16,
//   elems e -> K=khalf+e (= contiguous row chunk of Bt).
// ---------------------------------------------------------------------------
__device__ __forceinline__ v16bf frag_a(const __bf16* A, int lda, int k0, int lane) {
  int m  = lane & 15;
  int kb = (lane >> 4) << 4;
  const __bf16* p = A + (size_t)m * lda + k0 + kb;
  v16bf v;
  ((uint4*)&v)[0] = *(const uint4*)p;          // K = k0+kb+0..7
  ((uint4*)&v)[1] = *(const uint4*)(p + 8);    // K = k0+kb+8..15
  return v;
}
__device__ __forceinline__ v16bf frag_b(const __bf16* Bt, int ldb, int k0, int lane) {
  int n  = lane & 15;
  int kh = (lane >> 4) << 4;
  const __bf16* p = Bt + (size_t)n * ldb + k0 + kh;
  v16bf v;
  ((uint4*)&v)[0] = *(const uint4*)p;          // K = k0+kh+0..7
  ((uint4*)&v)[1] = *(const uint4*)(p + 8);    // K = k0+kh+8..15
  return v;
}

// ---------------------------------------------------------------------------
// Generic bf16 WMMA GEMM: Out(MxN) = act(A(MxK) * Bt(NxK)^T + bias) [+ resid]
// one 16x16 output tile per wave; exact grid => EXEC always full for WMMA.
// ---------------------------------------------------------------------------
__global__ void wmma_gemm_kernel(const __bf16* __restrict__ A,
                                 const __bf16* __restrict__ Bt,
                                 const float* __restrict__ bias,
                                 const float* __restrict__ resid,
                                 float* __restrict__ outF,
                                 __bf16* __restrict__ outB,
                                 int M, int N, int K, int doRelu) {
  int wid   = blockIdx.x * (blockDim.x >> 5) + (threadIdx.x >> 5);
  int lane  = threadIdx.x & 31;
  int tilesN = N >> 4;
  int tm = wid / tilesN, tn = wid - tm * tilesN;
  const __bf16* Ar = A  + (size_t)(tm << 4) * K;
  const __bf16* Br = Bt + (size_t)(tn << 4) * K;
  v8f acc = {0.f,0.f,0.f,0.f,0.f,0.f,0.f,0.f};
  for (int k0 = 0; k0 < K; k0 += 32) {
    v16bf a = frag_a(Ar, K, k0, lane);
    v16bf b = frag_b(Br, K, k0, lane);
    acc = __builtin_amdgcn_wmma_f32_16x16x32_bf16(false, a, false, b, (short)0, acc, false, false);
  }
  int n  = (tn << 4) + (lane & 15);
  int mh = (lane >> 4) << 3;
  float bv = bias ? bias[n] : 0.f;
#pragma unroll
  for (int v = 0; v < 8; ++v) {
    int m = (tm << 4) + mh + v;
    float x = acc[v] + bv;
    if (doRelu) x = fmaxf(x, 0.f);
    size_t o = (size_t)m * N + n;
    if (resid) x += resid[o];
    if (outF)  outF[o] = x;
    if (outB)  outB[o] = (__bf16)x;
  }
}

// ---------------------------------------------------------------------------
// Transition row normalizers: logZ[i] = logsumexp_j( r_state[i] . next[j] )
// Streams the 8192x8192 score matrix through WMMA with online logsumexp;
// never materializes it. All 8 A fragments (full K=256) are hoisted into
// registers (64 VGPRs), so the inner loop is pure B-stream + WMMA.
// ---------------------------------------------------------------------------
__global__ void trans_logz_kernel(const __bf16* __restrict__ Ab,
                                  const __bf16* __restrict__ Bb,
                                  float* __restrict__ logZ) {
  __shared__ float redM[16 * 128];
  __shared__ float redS[16 * 128];
  int tid  = threadIdx.x;
  int wave = tid >> 5, lane = tid & 31;
  int row0 = blockIdx.x << 4;
  const __bf16* Ar = Ab + (size_t)row0 * Hdim;

  v16bf af[8];
#pragma unroll
  for (int kk = 0; kk < 8; ++kk) af[kk] = frag_a(Ar, Hdim, kk << 5, lane);

  float rm[8], rs[8];
#pragma unroll
  for (int v = 0; v < 8; ++v) { rm[v] = NEGF; rs[v] = 0.f; }

  for (int jt = wave; jt < (CS >> 4); jt += 8) {
    v8f acc = {0.f,0.f,0.f,0.f,0.f,0.f,0.f,0.f};
    const __bf16* Br = Bb + (size_t)(jt << 4) * Hdim;
#pragma unroll
    for (int kk = 0; kk < 8; ++kk) {
      v16bf b = frag_b(Br, Hdim, kk << 5, lane);
      acc = __builtin_amdgcn_wmma_f32_16x16x32_bf16(false, af[kk], false, b, (short)0, acc, false, false);
    }
#pragma unroll
    for (int v = 0; v < 8; ++v) {
      float val = acc[v];
      float mn  = fmaxf(rm[v], val);
      rs[v] = rs[v] * __expf(rm[v] - mn) + __expf(val - mn);
      rm[v] = mn;
    }
  }
  int slot  = wave * 16 + (lane & 15);
  int rbase = (lane >> 4) << 3;
#pragma unroll
  for (int v = 0; v < 8; ++v) {
    redM[(rbase + v) * 128 + slot] = rm[v];
    redS[(rbase + v) * 128 + slot] = rs[v];
  }
  __syncthreads();
  if (tid < 16) {
    float M = NEGF;
    for (int k = 0; k < 128; ++k) M = fmaxf(M, redM[tid * 128 + k]);
    float S = 0.f;
    for (int k = 0; k < 128; ++k) S += redS[tid * 128 + k] * __expf(redM[tid * 128 + k] - M);
    logZ[row0 + tid] = M + __logf(S);
  }
}

// ---------------------------------------------------------------------------
// pots[b,t] = 64x64 block of normalized transition for (cluster(text[b,t]),
// cluster(text[b,t+1])), recomputed on demand with WMMA.
// ---------------------------------------------------------------------------
__global__ void pots_kernel(const int* __restrict__ text, const int* __restrict__ w2c,
                            const __bf16* __restrict__ rstateb,
                            const __bf16* __restrict__ nextb,
                            const float* __restrict__ logZt,
                            float* __restrict__ pots) {
  int b = blockIdx.x / (TT - 1);
  int t = blockIdx.x - b * (TT - 1);
  int c1 = w2c[text[b * TT + t]];
  int c2 = w2c[text[b * TT + t + 1]];
  int wave = threadIdx.x >> 5, lane = threadIdx.x & 31;
  const __bf16* A  = rstateb + (size_t)(c1 << 6) * Hdim;
  const __bf16* Bt = nextb   + (size_t)(c2 << 6) * Hdim;
  float* P = pots + (size_t)blockIdx.x * 64 * 64;
  for (int tile = wave; tile < 16; tile += 8) {
    int ti = tile >> 2, tj = tile & 3;
    v8f acc = {0.f,0.f,0.f,0.f,0.f,0.f,0.f,0.f};
    const __bf16* Ar = A  + (size_t)(ti << 4) * Hdim;
    const __bf16* Br = Bt + (size_t)(tj << 4) * Hdim;
#pragma unroll
    for (int kk = 0; kk < 8; ++kk) {
      v16bf a  = frag_a(Ar, Hdim, kk << 5, lane);
      v16bf bb = frag_b(Br, Hdim, kk << 5, lane);
      acc = __builtin_amdgcn_wmma_f32_16x16x32_bf16(false, a, false, bb, (short)0, acc, false, false);
    }
    int n  = (tj << 4) + (lane & 15);
    int mh = (lane >> 4) << 3;
#pragma unroll
    for (int v = 0; v < 8; ++v) {
      int m = (ti << 4) + mh + v;
      P[m * 64 + n] = acc[v] - logZt[(c1 << 6) + m];
    }
  }
}

// --------------------------- small helper kernels ---------------------------
__global__ void emb_kernel(const float* __restrict__ e1, const float* __restrict__ e2,
                           float* __restrict__ out) {
  int idx = blockIdx.x * blockDim.x + threadIdx.x;   // CS*H
  int h = idx & (Hdim - 1);
  int s = idx >> 8;
  out[idx] = e1[(s >> 6) * Hdim + h] + e2[(s & 63) * Hdim + h];
}

__global__ void cast_bf16_kernel(const float* __restrict__ in, __bf16* __restrict__ out, int n) {
  int i = blockIdx.x * blockDim.x + threadIdx.x;
  if (i < n) out[i] = (__bf16)in[i];
}

__global__ void transpose_wb16_kernel(const float* __restrict__ W, __bf16* __restrict__ Wt) {
  int idx = blockIdx.x * blockDim.x + threadIdx.x;   // H*H
  int r = idx >> 8, c = idx & 255;
  Wt[c * Hdim + r] = (__bf16)W[idx];
}

__global__ void transpose_term_kernel(const float* __restrict__ W, float* __restrict__ Wt) {
  int idx = blockIdx.x * blockDim.x + threadIdx.x;   // H*V
  if (idx >= Hdim * VV) return;
  int h = idx / VV, w = idx - h * VV;
  Wt[(size_t)w * Hdim + h] = W[idx];
}

__global__ void w2c_kernel(const int* __restrict__ word2state, int* __restrict__ w2c) {
  int w = blockIdx.x * blockDim.x + threadIdx.x;
  if (w < VV) w2c[w] = word2state[w * SPWc] >> 6;
}

// E[w,i] = r_pre[cluster(w)*64+i] . term_pwT[w] + term_pb[w]
__global__ void E_kernel(const float* __restrict__ rpre, const float* __restrict__ termT,
                         const float* __restrict__ termPb, const int* __restrict__ w2c,
                         float* __restrict__ E) {
  int w = blockIdx.x, i = threadIdx.x;               // blockDim = 64
  __shared__ float col[Hdim];
  for (int k = i; k < Hdim; k += 64) col[k] = termT[(size_t)w * Hdim + k];
  __syncthreads();
  const float* r = rpre + (size_t)((w2c[w] << 6) + i) * Hdim;
  float acc = 0.f;
#pragma unroll 8
  for (int k = 0; k < Hdim; ++k) acc += r[k] * col[k];
  E[w * 64 + i] = acc + termPb[w];
}

// logZe[s] = logsumexp over {w : cluster(w)==s>>6} of E[w, s&63]
__global__ void elogz_kernel(const float* __restrict__ E, const int* __restrict__ w2c,
                             float* __restrict__ logZe) {
  int s = blockIdx.x;
  int c = s >> 6, i = s & 63;
  __shared__ float red[256];
  int tid = threadIdx.x;
  float M = NEGF;
  for (int w = tid; w < VV; w += 256)
    if (w2c[w] == c) M = fmaxf(M, E[w * 64 + i]);
  red[tid] = M; __syncthreads();
  for (int st = 128; st > 0; st >>= 1) { if (tid < st) red[tid] = fmaxf(red[tid], red[tid + st]); __syncthreads(); }
  M = red[0]; __syncthreads();
  float S = 0.f;
  for (int w = tid; w < VV; w += 256)
    if (w2c[w] == c) S += __expf(E[w * 64 + i] - M);
  red[tid] = S; __syncthreads();
  for (int st = 128; st > 0; st >>= 1) { if (tid < st) red[tid] += red[tid + st]; __syncthreads(); }
  if (tid == 0) logZe[s] = (red[0] > 0.f) ? (M + __logf(red[0])) : NEGF;
}

__global__ void obs_kernel(const int* __restrict__ text, const int* __restrict__ w2c,
                           const float* __restrict__ E, const float* __restrict__ logZe,
                           float* __restrict__ obs) {
  int idx = blockIdx.x * blockDim.x + threadIdx.x;   // B*T*64
  int i = idx & 63, bt = idx >> 6;
  int w = text[bt], c = w2c[w];
  obs[idx] = E[w * 64 + i] - logZe[(c << 6) + i];
}

__global__ void sc_kernel(const float* __restrict__ rstart, const float* __restrict__ pw,
                          const float* __restrict__ pb, float* __restrict__ sc) {
  int s = blockIdx.x * blockDim.x + threadIdx.x;     // CS
  const float* r = rstart + (size_t)s * Hdim;
  float a = 0.f;
#pragma unroll 8
  for (int k = 0; k < Hdim; ++k) a += r[k] * pw[k];
  sc[s] = a + pb[0];
}

__global__ void lse_kernel(const float* __restrict__ sc, float* __restrict__ lse) {
  __shared__ float red[256];
  int tid = threadIdx.x;
  float M = NEGF;
  for (int i = tid; i < CS; i += 256) M = fmaxf(M, sc[i]);
  red[tid] = M; __syncthreads();
  for (int st = 128; st > 0; st >>= 1) { if (tid < st) red[tid] = fmaxf(red[tid], red[tid + st]); __syncthreads(); }
  M = red[0]; __syncthreads();
  float S = 0.f;
  for (int i = tid; i < CS; i += 256) S += __expf(sc[i] - M);
  red[tid] = S; __syncthreads();
  for (int st = 128; st > 0; st >>= 1) { if (tid < st) red[tid] += red[tid + st]; __syncthreads(); }
  if (tid == 0) lse[0] = M + __logf(red[0]);
}

// Forward scan: one block (64 threads) per batch element.
__global__ void scan_kernel(const int* __restrict__ text, const int* __restrict__ w2c,
                            const float* __restrict__ sc, const float* __restrict__ lse,
                            const float* __restrict__ obs, const float* __restrict__ pots,
                            float* __restrict__ out) {
  int b = blockIdx.x, j = threadIdx.x;               // blockDim = 64
  __shared__ float sh[64];
  __shared__ float red[64];
  int c0 = w2c[text[b * TT]];
  float alpha = sc[(c0 << 6) + j] - lse[0] + obs[(size_t)(b * TT) * 64 + j];
  for (int t = 0; t < TT - 1; ++t) {
    sh[j] = alpha;
    __syncthreads();
    const float* P = pots + (size_t)(b * (TT - 1) + t) * 64 * 64 + j;
    float M = NEGF;
#pragma unroll 4
    for (int i = 0; i < 64; ++i) M = fmaxf(M, sh[i] + P[i * 64]);
    float S = 0.f;
#pragma unroll 4
    for (int i = 0; i < 64; ++i) S += __expf(sh[i] + P[i * 64] - M);
    float a2 = M + __logf(S) + obs[(size_t)(b * TT + t + 1) * 64 + j];
    __syncthreads();
    alpha = a2;
  }
  red[j] = alpha; __syncthreads();
  for (int st = 32; st > 0; st >>= 1) { if (j < st) red[j] = fmaxf(red[j], red[j + st]); __syncthreads(); }
  float M = red[0]; __syncthreads();
  red[j] = __expf(alpha - M); __syncthreads();
  for (int st = 32; st > 0; st >>= 1) { if (j < st) red[j] += red[j + st]; __syncthreads(); }
  if (j == 0) out[b] = M + __logf(red[0]);
}

// ---------------------------------------------------------------------------
extern "C" void kernel_launch(void* const* d_in, const int* in_sizes, int n_in,
                              void* d_out, int out_size, void* d_ws, size_t ws_size,
                              hipStream_t stream) {
  (void)in_sizes; (void)n_in; (void)out_size; (void)ws_size;
  const int*   text       = (const int*)d_in[0];
  const int*   word2state = (const int*)d_in[1];
  const float* se1_start  = (const float*)d_in[2];
  const float* se2_start  = (const float*)d_in[3];
  const float* start_w1   = (const float*)d_in[4];
  const float* start_b1   = (const float*)d_in[5];
  const float* start_w2   = (const float*)d_in[6];
  const float* start_b2   = (const float*)d_in[7];
  const float* start_pw   = (const float*)d_in[8];
  const float* start_pb   = (const float*)d_in[9];
  const float* se1_state  = (const float*)d_in[10];
  const float* se2_state  = (const float*)d_in[11];
  const float* se1_next   = (const float*)d_in[12];
  const float* se2_next   = (const float*)d_in[13];
  const float* trans_w1   = (const float*)d_in[14];
  const float* trans_b1   = (const float*)d_in[15];
  const float* trans_w2   = (const float*)d_in[16];
  const float* trans_b2   = (const float*)d_in[17];
  const float* se1_pre    = (const float*)d_in[18];
  const float* se2_pre    = (const float*)d_in[19];
  const float* term_w1    = (const float*)d_in[20];
  const float* term_b1    = (const float*)d_in[21];
  const float* term_w2    = (const float*)d_in[22];
  const float* term_b2    = (const float*)d_in[23];
  const float* term_pw    = (const float*)d_in[24];
  const float* term_pb    = (const float*)d_in[25];
  float* out = (float*)d_out;

  // ---- workspace carve-up (256B aligned) ----
  char* p = (char*)d_ws;
  auto alloc = [&](size_t bytes) -> void* {
    void* r = (void*)p;
    p += (bytes + 255) & ~(size_t)255;
    return r;
  };
  const size_t CH = (size_t)CS * Hdim;
  float*  start_emb = (float*)alloc(CH * 4);
  float*  state_emb = (float*)alloc(CH * 4);
  float*  next_emb  = (float*)alloc(CH * 4);
  float*  pre_emb   = (float*)alloc(CH * 4);
  float*  r_start   = (float*)alloc(CH * 4);
  float*  r_pre     = (float*)alloc(CH * 4);
  float*  termT     = (float*)alloc((size_t)VV * Hdim * 4);
  float*  Etab      = (float*)alloc((size_t)VV * 64 * 4);
  float*  pots      = (float*)alloc((size_t)BB * (TT - 1) * 64 * 64 * 4);
  float*  obs       = (float*)alloc((size_t)BB * TT * 64 * 4);
  float*  sc        = (float*)alloc(CS * 4);
  float*  logZt     = (float*)alloc(CS * 4);
  float*  logZe     = (float*)alloc(CS * 4);
  float*  lse       = (float*)alloc(256);
  __bf16* xb        = (__bf16*)alloc(CH * 2);
  __bf16* hb        = (__bf16*)alloc(CH * 2);
  __bf16* rstateb   = (__bf16*)alloc(CH * 2);
  __bf16* nextb     = (__bf16*)alloc(CH * 2);
  __bf16* w1t       = (__bf16*)alloc((size_t)Hdim * Hdim * 2);
  __bf16* w2t       = (__bf16*)alloc((size_t)Hdim * Hdim * 2);
  int*    w2c       = (int*)alloc(VV * 4);

  const int gemmBlocks = (CS / 16) * (Hdim / 16) / 8;   // 1024 blocks, 8 waves each

  // word -> cluster map
  w2c_kernel<<<(VV + 255) / 256, 256, 0, stream>>>(word2state, w2c);

  // embeddings (C x H each)
  emb_kernel<<<CH / 256, 256, 0, stream>>>(se1_start, se2_start, start_emb);
  emb_kernel<<<CH / 256, 256, 0, stream>>>(se1_state, se2_state, state_emb);
  emb_kernel<<<CH / 256, 256, 0, stream>>>(se1_next,  se2_next,  next_emb);
  emb_kernel<<<CH / 256, 256, 0, stream>>>(se1_pre,   se2_pre,   pre_emb);

  // residual(start_emb) -> r_start (f32)
  cast_bf16_kernel<<<CH / 256, 256, 0, stream>>>(start_emb, xb, (int)CH);
  transpose_wb16_kernel<<<Hdim * Hdim / 256, 256, 0, stream>>>(start_w1, w1t);
  transpose_wb16_kernel<<<Hdim * Hdim / 256, 256, 0, stream>>>(start_w2, w2t);
  wmma_gemm_kernel<<<gemmBlocks, 256, 0, stream>>>(xb, w1t, start_b1, nullptr, nullptr, hb, CS, Hdim, Hdim, 1);
  wmma_gemm_kernel<<<gemmBlocks, 256, 0, stream>>>(hb, w2t, start_b2, start_emb, r_start, nullptr, CS, Hdim, Hdim, 1);

  // residual(state_emb) -> rstateb (bf16 only: A operand of transition GEMMs)
  cast_bf16_kernel<<<CH / 256, 256, 0, stream>>>(state_emb, xb, (int)CH);
  transpose_wb16_kernel<<<Hdim * Hdim / 256, 256, 0, stream>>>(trans_w1, w1t);
  transpose_wb16_kernel<<<Hdim * Hdim / 256, 256, 0, stream>>>(trans_w2, w2t);
  wmma_gemm_kernel<<<gemmBlocks, 256, 0, stream>>>(xb, w1t, trans_b1, nullptr, nullptr, hb, CS, Hdim, Hdim, 1);
  wmma_gemm_kernel<<<gemmBlocks, 256, 0, stream>>>(hb, w2t, trans_b2, state_emb, nullptr, rstateb, CS, Hdim, Hdim, 1);

  // next_emb bf16 (B operand)
  cast_bf16_kernel<<<CH / 256, 256, 0, stream>>>(next_emb, nextb, (int)CH);

  // residual(pre_emb) -> r_pre (f32, feeds E table dots)
  cast_bf16_kernel<<<CH / 256, 256, 0, stream>>>(pre_emb, xb, (int)CH);
  transpose_wb16_kernel<<<Hdim * Hdim / 256, 256, 0, stream>>>(term_w1, w1t);
  transpose_wb16_kernel<<<Hdim * Hdim / 256, 256, 0, stream>>>(term_w2, w2t);
  wmma_gemm_kernel<<<gemmBlocks, 256, 0, stream>>>(xb, w1t, term_b1, nullptr, nullptr, hb, CS, Hdim, Hdim, 1);
  wmma_gemm_kernel<<<gemmBlocks, 256, 0, stream>>>(hb, w2t, term_b2, pre_emb, r_pre, nullptr, CS, Hdim, Hdim, 1);

  // transition row normalizers (streaming 8192x8192 WMMA, no materialization)
  trans_logz_kernel<<<CS / 16, 256, 0, stream>>>(rstateb, nextb, logZt);

  // on-demand 64x64 transition blocks for each (b,t)
  pots_kernel<<<BB * (TT - 1), 256, 0, stream>>>(text, w2c, rstateb, nextb, logZt, pots);

  // emission: only masked-true entries + per-cluster normalizers
  transpose_term_kernel<<<(Hdim * VV + 255) / 256, 256, 0, stream>>>(term_pw, termT);
  E_kernel<<<VV, 64, 0, stream>>>(r_pre, termT, term_pb, w2c, Etab);
  elogz_kernel<<<CS, 256, 0, stream>>>(Etab, w2c, logZe);
  obs_kernel<<<BB * TT * 64 / 256, 256, 0, stream>>>(text, w2c, Etab, logZe, obs);

  // start distribution
  sc_kernel<<<CS / 256, 256, 0, stream>>>(r_start, start_pw, start_pb, sc);
  lse_kernel<<<1, 256, 0, stream>>>(sc, lse);

  // forward scan + final logsumexp
  scan_kernel<<<BB, 64, 0, stream>>>(text, w2c, sc, lse, obs, pots, out);
}
